// FasterRCNN_51007031608004
// MI455X (gfx1250) — matrix operand order repaired and checked by reading, not compiled, not run on hardware
//
#include <hip/hip_runtime.h>

// ---------------------------------------------------------------------------
// Types for CDNA5 WMMA (wave32, v_wmma_f32_16x16x32_bf16)
// ---------------------------------------------------------------------------
typedef __attribute__((ext_vector_type(8)))  __bf16 bf16x8;
typedef __attribute__((ext_vector_type(16))) __bf16 bf16x16;
typedef __attribute__((ext_vector_type(8)))  float  f32x8;
typedef __attribute__((ext_vector_type(4)))  int    i32x4;

#define NROI   1024     // BS(2) * R(512)
#define RPB    512
#define CCH    256
#define FDIM   12544    // 256*7*7
#define DDIM   1024
#define NCLS_R 81
#define NCLS_P 128      // padded to one BN tile
#define NREG   320      // 80*4
#define NREG_P 384      // padded to 3 BN tiles

// GEMM tiling
#define BM  128
#define BN  128
#define BK  64
#define LDK (BK + 8)    // padded LDS row stride (bf16 elems) -> conflict-free frags

// ---------------------------------------------------------------------------
// CDNA5 async global->LDS path (ASYNCcnt-tracked).
// Probe-confirmed signature (from the round-2 diagnostic): param0 is
// 'int4 * addrspace(1)' (non-const), so: (global int4*, lds int4*, imm, imm).
// ---------------------------------------------------------------------------
#if defined(__has_builtin)
#if __has_builtin(__builtin_amdgcn_global_load_async_to_lds_b128)
#define ASYNC_LDS 1
#endif
#endif

#if defined(ASYNC_LDS)
#if __has_builtin(__builtin_amdgcn_s_wait_asynccnt)
#define WAIT_ASYNC(n) __builtin_amdgcn_s_wait_asynccnt((short)(n))
#else
#define WAIT_ASYNC(n) asm volatile("s_wait_asynccnt %0" ::"i"(n) : "memory")
#endif
typedef __attribute__((address_space(1))) i32x4* as1_v4i;
typedef __attribute__((address_space(3))) i32x4* as3_v4i;
__device__ __forceinline__ void async_cp16(const void* g, void* l) {
    __builtin_amdgcn_global_load_async_to_lds_b128(
        (as1_v4i)(void*)g, (as3_v4i)l, 0, 0);
}
#else
#define WAIT_ASYNC(n)
__device__ __forceinline__ void async_cp16(const void* g, void* l) {
    *(bf16x8*)l = *(const bf16x8*)g;   // sync fallback: global_load + ds_store
}
#endif

// ---------------------------------------------------------------------------
// fp32 -> bf16 conversion with zero row-padding (padding is trailing rows,
// flat indexing works because K is never padded)
// ---------------------------------------------------------------------------
__global__ void cvt_bf16_kernel(const float* __restrict__ src,
                                __bf16* __restrict__ dst,
                                size_t srcCount, size_t dstCount) {
    size_t i = (size_t)blockIdx.x * blockDim.x + threadIdx.x;
    size_t stride = (size_t)gridDim.x * blockDim.x;
    for (; i < dstCount; i += stride) {
        float v = (i < srcCount) ? src[i] : 0.0f;
        dst[i] = (__bf16)v;
    }
}

// ---------------------------------------------------------------------------
// ROI-Align with FPN level routing (aligned=False, SR=2, OUT=7).
// One block per ROI, one thread per channel. Writes A-matrix [NROI, FDIM] bf16
// with feature index = c*49 + oy*7 + ox (matches reshape(N, C,7,7)->(N,-1)).
// ---------------------------------------------------------------------------
__global__ void roi_align_kernel(const float* __restrict__ f2,
                                 const float* __restrict__ f3,
                                 const float* __restrict__ f4,
                                 const float* __restrict__ f5,
                                 const float* __restrict__ rois,
                                 __bf16* __restrict__ outA) {
    const int n = blockIdx.x;      // roi index
    const int c = threadIdx.x;     // channel
    const int b = n / RPB;

    const float x1 = rois[n * 4 + 0], y1 = rois[n * 4 + 1];
    const float x2 = rois[n * 4 + 2], y2 = rois[n * 4 + 3];
    const float bw_ = x2 - x1, bh_ = y2 - y1;
    const float lvf = floorf(4.0f + log2f(sqrtf(bw_ * bh_) / 224.0f + 1e-6f));
    const int   li  = (int)lvf;

    const float* feat = nullptr; int H = 0; float scale = 0.0f;
    if      (li == 2) { feat = f2; H = 200; scale = 0.25f;    }
    else if (li == 3) { feat = f3; H = 100; scale = 0.125f;   }
    else if (li == 4) { feat = f4; H = 50;  scale = 0.0625f;  }
    else if (li == 5) { feat = f5; H = 25;  scale = 0.03125f; }

    __bf16* dst = outA + (size_t)n * FDIM + (size_t)c * 49;

    if (feat == nullptr) {  // routed outside the pyramid -> zeros
        for (int bin = 0; bin < 49; ++bin) dst[bin] = (__bf16)0.0f;
        return;
    }

    const int   W   = H;
    const float xs1 = x1 * scale, ys1 = y1 * scale;
    const float rw  = fmaxf(x2 * scale - xs1, 1.0f);
    const float rh  = fmaxf(y2 * scale - ys1, 1.0f);
    const float binw = rw / 7.0f, binh = rh / 7.0f;
    const float* fbase = feat + ((size_t)b * CCH + c) * (size_t)H * W;

    for (int bin = 0; bin < 49; ++bin) {
        const int oy = bin / 7, ox = bin % 7;
        float acc = 0.0f;
#pragma unroll
        for (int sy = 0; sy < 2; ++sy) {
#pragma unroll
            for (int sx = 0; sx < 2; ++sx) {
                float y = ys1 + ((float)oy + ((float)sy + 0.5f) * 0.5f) * binh;
                float x = xs1 + ((float)ox + ((float)sx + 0.5f) * 0.5f) * binw;
                if (y > -1.0f && y < (float)H && x > -1.0f && x < (float)W) {
                    y = fminf(fmaxf(y, 0.0f), (float)(H - 1));
                    x = fminf(fmaxf(x, 0.0f), (float)(W - 1));
                    const float y0f = floorf(y), x0f = floorf(x);
                    const int y0 = (int)y0f, x0 = (int)x0f;
                    const int yb = min(y0 + 1, H - 1), xb = min(x0 + 1, W - 1);
                    const float ly = y - y0f, lx = x - x0f;
                    const float v00 = fbase[y0 * W + x0];
                    const float v01 = fbase[y0 * W + xb];
                    const float v10 = fbase[yb * W + x0];
                    const float v11 = fbase[yb * W + xb];
                    acc += (1.0f - ly) * (1.0f - lx) * v00
                         + (1.0f - ly) * lx          * v01
                         + ly          * (1.0f - lx) * v10
                         + ly          * lx          * v11;
                }
            }
        }
        dst[bin] = (__bf16)(acc * 0.25f);
    }
}

// ---------------------------------------------------------------------------
// bf16 WMMA GEMM with LDS tiling + async global->LDS double buffering.
//   out[M,N] = act( A[M,K] @ W[N,K]^T + bias[N] )
// Workgroup (8 waves) computes a 128x128 tile; wave (wm, wn) computes 64x32
// = 4x2 WMMA fragments. LDS tiles A[128][BK], B[128][BK], double buffered,
// rows padded by 8 bf16 so fragment reads hit 16 distinct banks.
// Fragment packing (this code's contiguous k order, same for A and B):
//   A (16x32): lane L -> m = L%16, half=L/16; elems 0..15 = K[half*16..+15]
//   B (32x16): lane L -> n = L%16, half=L/16; elems 0..15 = K[half*16..+15]
//   C/D f32  : lane L, elem r -> m = 8*(L/16)+r, n = L%16
// ---------------------------------------------------------------------------
__device__ __forceinline__ bf16x16 pack16(bf16x8 lo, bf16x8 hi) {
    return __builtin_shufflevector(lo, hi, 0, 1, 2, 3, 4, 5, 6, 7,
                                           8, 9, 10, 11, 12, 13, 14, 15);
}

__global__ __launch_bounds__(256)
void wmma_gemm_lds_kernel(const __bf16* __restrict__ A,
                          const __bf16* __restrict__ Wt,
                          const float* __restrict__ bias,
                          float*  __restrict__ outF,
                          __bf16* __restrict__ outB,
                          int N_real, int K, int ldout, int relu) {
    __shared__ __bf16 As[2][BM][LDK];
    __shared__ __bf16 Bs[2][BN][LDK];

    const int tid  = threadIdx.x;
    const int lane = tid & 31;
    const int wave = tid >> 5;          // 0..7
    const int wm   = wave & 1;          // M half (64 rows)
    const int wn   = wave >> 1;         // N quarter (32 cols)
    const int half = lane >> 4;
    const int l16  = lane & 15;
    const int m0   = blockIdx.y * BM;
    const int n0   = blockIdx.x * BN;

    f32x8 acc[4][2] = {};

    // Loader: tile is BM x BK bf16 = 1024 chunks of 16B; 4 chunks per thread.
    // chunk ch -> row = ch/8, col = (ch%8)*8 elems.
    auto issue = [&](int buf, int k) {
#pragma unroll
        for (int c = 0; c < 4; ++c) {
            const int ch  = tid + c * 256;
            const int row = ch >> 3;
            const int col = (ch & 7) * 8;
            async_cp16(A  + (size_t)(m0 + row) * K + k + col, &As[buf][row][col]);
            async_cp16(Wt + (size_t)(n0 + row) * K + k + col, &Bs[buf][row][col]);
        }
    };

    const int nk = K / BK;
    issue(0, 0);

    for (int i = 0; i < nk; ++i) {
        const int cur = i & 1;
        const bool more = (i + 1) < nk;
        if (more) issue(cur ^ 1, (i + 1) * BK);   // overlap next tile's loads
        if (more) { WAIT_ASYNC(8); } else { WAIT_ASYNC(0); }
        __syncthreads();                           // cur tile visible to all waves

#pragma unroll
        for (int ks = 0; ks < BK / 32; ++ks) {
            bf16x16 af[4];
#pragma unroll
            for (int mt = 0; mt < 4; ++mt) {
                const __bf16* p = &As[cur][wm * 64 + mt * 16 + l16][ks * 32 + half * 16];
                af[mt] = pack16(*(const bf16x8*)p, *(const bf16x8*)(p + 8));
            }
            bf16x16 bfr[2];
#pragma unroll
            for (int nt = 0; nt < 2; ++nt) {
                const __bf16* p = &Bs[cur][wn * 32 + nt * 16 + l16][ks * 32 + half * 16];
                bfr[nt] = pack16(*(const bf16x8*)p, *(const bf16x8*)(p + 8));
            }
#pragma unroll
            for (int mt = 0; mt < 4; ++mt)
#pragma unroll
                for (int nt = 0; nt < 2; ++nt)
                    acc[mt][nt] = __builtin_amdgcn_wmma_f32_16x16x32_bf16(
                        false, af[mt], false, bfr[nt], (short)0, acc[mt][nt],
                        false, false);
        }
        __syncthreads();   // all reads of cur done before it is refilled
    }

    // epilogue: bias + optional ReLU, store bf16 (intermediate) or f32 (heads)
#pragma unroll
    for (int mt = 0; mt < 4; ++mt) {
#pragma unroll
        for (int nt = 0; nt < 2; ++nt) {
            const int n  = n0 + wn * 32 + nt * 16 + l16;
            const float bv = (n < N_real) ? bias[n] : 0.0f;
#pragma unroll
            for (int r = 0; r < 8; ++r) {
                const int m = m0 + wm * 64 + mt * 16 + half * 8 + r;
                float v = acc[mt][nt][r] + bv;
                if (relu) v = fmaxf(v, 0.0f);
                if (outB) {
                    outB[(size_t)m * ldout + n] = (__bf16)v;
                } else if (n < N_real) {
                    outF[(size_t)m * ldout + n] = v;
                }
            }
        }
    }
}

// ---------------------------------------------------------------------------
// Host-side orchestration
// ---------------------------------------------------------------------------
extern "C" void kernel_launch(void* const* d_in, const int* in_sizes, int n_in,
                              void* d_out, int out_size, void* d_ws, size_t ws_size,
                              hipStream_t stream) {
    (void)in_sizes; (void)n_in; (void)out_size; (void)ws_size;

    const float* feat2 = (const float*)d_in[0];
    const float* feat3 = (const float*)d_in[1];
    const float* feat4 = (const float*)d_in[2];
    const float* feat5 = (const float*)d_in[3];
    const float* rois  = (const float*)d_in[4];
    const float* W_fc1 = (const float*)d_in[5];
    const float* b_fc1 = (const float*)d_in[6];
    const float* W_fc2 = (const float*)d_in[7];
    const float* b_fc2 = (const float*)d_in[8];
    const float* W_cls = (const float*)d_in[9];
    const float* b_cls = (const float*)d_in[10];
    const float* W_reg = (const float*)d_in[11];
    const float* b_reg = (const float*)d_in[12];

    // workspace layout (bf16 elements, all 256B-aligned sizes)
    char* ws = (char*)d_ws;
    __bf16* roiA  = (__bf16*)ws;  ws += (size_t)NROI * FDIM * 2;
    __bf16* Wfc1b = (__bf16*)ws;  ws += (size_t)DDIM * FDIM * 2;
    __bf16* Wfc2b = (__bf16*)ws;  ws += (size_t)DDIM * DDIM * 2;
    __bf16* Wclsb = (__bf16*)ws;  ws += (size_t)NCLS_P * DDIM * 2;
    __bf16* Wregb = (__bf16*)ws;  ws += (size_t)NREG_P * DDIM * 2;
    __bf16* x1    = (__bf16*)ws;  ws += (size_t)NROI * DDIM * 2;
    __bf16* x2    = (__bf16*)ws;  ws += (size_t)NROI * DDIM * 2;

    float* outCls = (float*)d_out;                   // [1024, 81]
    float* outReg = outCls + (size_t)NROI * NCLS_R;  // [1024, 320]

    // 1) weight conversion / zero-padding to bf16 (pads rows to BN multiples)
    cvt_bf16_kernel<<<4096, 256, 0, stream>>>(W_fc1, Wfc1b,
        (size_t)DDIM * FDIM, (size_t)DDIM * FDIM);
    cvt_bf16_kernel<<<1024, 256, 0, stream>>>(W_fc2, Wfc2b,
        (size_t)DDIM * DDIM, (size_t)DDIM * DDIM);
    cvt_bf16_kernel<<<128, 256, 0, stream>>>(W_cls, Wclsb,
        (size_t)NCLS_R * DDIM, (size_t)NCLS_P * DDIM);
    cvt_bf16_kernel<<<256, 256, 0, stream>>>(W_reg, Wregb,
        (size_t)NREG * DDIM, (size_t)NREG_P * DDIM);

    // 2) ROI-align -> bf16 A matrix [1024, 12544]
    roi_align_kernel<<<NROI, CCH, 0, stream>>>(feat2, feat3, feat4, feat5, rois, roiA);

    // 3) FC1: [1024,12544]x[12544,1024] + bias, ReLU -> x1 (bf16)
    wmma_gemm_lds_kernel<<<dim3(DDIM / BN, NROI / BM), 256, 0, stream>>>(
        roiA, Wfc1b, b_fc1, nullptr, x1, DDIM, FDIM, DDIM, 1);

    // 4) FC2: [1024,1024]^2 + bias, ReLU -> x2 (bf16)
    wmma_gemm_lds_kernel<<<dim3(DDIM / BN, NROI / BM), 256, 0, stream>>>(
        x1, Wfc2b, b_fc2, nullptr, x2, DDIM, DDIM, DDIM, 1);

    // 5) cls head: [1024,1024]x[1024,81] -> f32 out (weights padded to 128)
    wmma_gemm_lds_kernel<<<dim3(NCLS_P / BN, NROI / BM), 256, 0, stream>>>(
        x2, Wclsb, b_cls, outCls, nullptr, NCLS_R, DDIM, NCLS_R, 0);

    // 6) reg head: [1024,1024]x[1024,320] -> f32 out (weights padded to 384)
    wmma_gemm_lds_kernel<<<dim3(NREG_P / BN, NROI / BM), 256, 0, stream>>>(
        x2, Wregb, b_reg, outReg, nullptr, NREG, DDIM, NREG, 0);
}
